// MoeEPOnly_89292370084490
// MI455X (gfx1250) — compile-verified, matrix operand order checked by reading, not run on hardware
//
#include <hip/hip_runtime.h>

// ---------------- problem constants ----------------
#define E_NUM   16
#define TOPKK   2
#define NTOK    8192          // B*S = 4*2048
#define HDIM    1024
#define IDIM    512
#define ISDIM   2048
#define BMROWS  128           // block row tile (expert segments padded to this)
#define MCAP    (NTOK*TOPKK + E_NUM*BMROWS)   // 18432 padded expert rows
#define MAXTILES (MCAP/BMROWS)                // 144

typedef __attribute__((ext_vector_type(16))) __bf16        v16bf;
typedef __attribute__((ext_vector_type(8)))  float         v8f;
typedef __attribute__((ext_vector_type(4)))  unsigned int  v4u;
typedef __attribute__((ext_vector_type(8)))  unsigned int  v8u;
typedef __attribute__((ext_vector_type(4)))  float         v4f;

typedef __attribute__((address_space(3))) void* as3_void_ptr;

#if defined(__gfx1250__)
#define HAS_ASYNC_LDS 1
#else
#define HAS_ASYNC_LDS 0
#endif

#if __has_builtin(__builtin_amdgcn_s_wait_asynccnt)
#define WAIT_ASYNC(n) __builtin_amdgcn_s_wait_asynccnt(n)
#else
#define WAIT_ASYNC(n) asm volatile("s_wait_asynccnt " #n ::: "memory")
#endif

__device__ __forceinline__ unsigned short f2bf(float f) {
  unsigned u = __builtin_bit_cast(unsigned, f);
  u += 0x7FFFu + ((u >> 16) & 1u);          // round-to-nearest-even
  return (unsigned short)(u >> 16);
}
__device__ __forceinline__ float bf2f(unsigned short h) {
  return __builtin_bit_cast(float, ((unsigned)h) << 16);
}
__device__ __forceinline__ float silu_f(float x) {
  return x * (1.0f / (1.0f + __expf(-x)));
}

#if HAS_ASYNC_LDS
// ASYNCcnt-tracked memory->LDS copy, 16 bytes per lane (VGLOBAL encoding:
// VDST = LDS byte address VGPR, VADDR = 64-bit global address VGPR pair).
__device__ __forceinline__ void async_cp16(const void* g, void* l) {
  unsigned lds_off = (unsigned)(unsigned long long)(as3_void_ptr)l;
  asm volatile("global_load_async_to_lds_b128 %0, %1, off"
               :: "v"(lds_off), "v"(g) : "memory");
}
#endif

// One 16x32 bf16 fragment (A/B operand layout per CDNA5 ISA 7.12.2):
// lane L: M/N = L%16, half = L/16; VGPRs 0..3 = K half*8..half*8+7 (one
// ds_load_b128), VGPRs 4..7 = K+16 (second ds_load_b128).
__device__ __forceinline__ v16bf load_frag(const unsigned short* tile /*[rows][32]*/,
                                           int sub, int lane) {
  int m  = sub * 16 + (lane & 15);
  int kb = (lane >> 4) * 8;
  const v4u* p = (const v4u*)(tile + m * 32 + kb);
  v4u lo = p[0];      // K = kb .. kb+7
  v4u hi = p[2];      // K = kb+16 .. kb+23
  v8u f;
  f[0]=lo[0]; f[1]=lo[1]; f[2]=lo[2]; f[3]=lo[3];
  f[4]=hi[0]; f[5]=hi[1]; f[6]=hi[2]; f[7]=hi[3];
  return __builtin_bit_cast(v16bf, f);
}

// ---------------- generic bf16 WMMA GEMM, 128x64 block tile ----------------
// C[M,N] = A[M,K](bf16) * B[N,K](bf16)^T.  256 thr = 8 waves; wave owns a
// 2x2 grid of 16x16 f32 accumulators (4 WMMA per K-step, 8 ds_load_b128).
// MODE 0: linear rows, store bf16
// MODE 1: linear rows, store f32 * rowscale[row]     (initializes d_out)
// MODE 2: A rows gathered via perm, per-tile expert B, store bf16
// MODE 3: linear A rows, per-tile expert B, atomicAdd f32 out[perm[row]]*permw[row]
template<int MODE>
__global__ void __launch_bounds__(256)
gemm_bf16(const unsigned short* __restrict__ A,
          const unsigned short* __restrict__ B,
          void* __restrict__ Cout,
          int N, int K,
          const int*   __restrict__ perm,
          const float* __restrict__ permw,
          const int*   __restrict__ po,        // padded offsets [E_NUM+1]
          const float* __restrict__ rowscale,
          const unsigned short* __restrict__ zrow,  // 128 zero bf16
          long bStrideExpert)
{
  __shared__ __align__(16) unsigned short At[2][BMROWS * 32]; // 2 x 8KB
  __shared__ __align__(16) unsigned short Bt[2][64 * 32];     // 2 x 4KB
  __shared__ int s_e;

  const int tid   = threadIdx.x;
  const int lane  = tid & 31;
  const int wave  = tid >> 5;
  const int rsub0 = (wave & 3) * 2;    // 2 of 8 row sub-tiles
  const int csub0 = (wave >> 2) * 2;   // 2 of 4 col sub-tiles

  const int rowBase = blockIdx.x * BMROWS;
  const int colBase = blockIdx.y * 64;

  const unsigned short* Bp = B;
  if (MODE == 2 || MODE == 3) {
    if (tid == 0) {
      int t0 = rowBase;
      int e  = -1;
      if (t0 < po[E_NUM]) {
        e = 0;
        while (t0 >= po[e + 1]) e++;
      }
      s_e = e;
    }
    __syncthreads();
    int e = s_e;
    if (e < 0) return;
    Bp = B + (long)e * bStrideExpert;
  }

  // staging split: A tile 128x32 -> 2 b128 chunks/thread, B tile 64x32 -> 1
  const int rA   = tid >> 1;
  const int segA = (tid & 1) * 8;       // + second chunk at +16
  const int rB   = tid >> 2;
  const int segB = (tid & 3) * 8;

  const unsigned short* aRow;
  bool aDead = false;
  if (MODE == 2) {
    int tok = perm[rowBase + rA];
    aDead = (tok < 0);
    aRow  = aDead ? zrow : (A + (long)tok * K);
  } else {
    aRow = A + (long)(rowBase + rA) * K;
  }
  const unsigned short* bRow = Bp + (long)(colBase + rB) * K;

  v8f acc00 = {}, acc01 = {}, acc10 = {}, acc11 = {};

  const int nb = K >> 5;

#if HAS_ASYNC_LDS
  // ---------------- async double-buffered pipeline ----------------
  {
    const unsigned short* ga = aDead ? zrow : (aRow + segA);
    async_cp16(ga,                        &At[0][rA * 32 + segA]);
    async_cp16(aDead ? zrow : (ga + 16),  &At[0][rA * 32 + segA + 16]);
    async_cp16(bRow + segB,               &Bt[0][rB * 32 + segB]);
  }
  for (int i = 0; i < nb; i++) {
    const int cur = i & 1;
    if (i + 1 < nb) {
      int k0 = (i + 1) << 5;
      const unsigned short* ga = aDead ? zrow : (aRow + k0 + segA);
      async_cp16(ga,                        &At[cur ^ 1][rA * 32 + segA]);
      async_cp16(aDead ? zrow : (ga + 16),  &At[cur ^ 1][rA * 32 + segA + 16]);
      async_cp16(bRow + k0 + segB,          &Bt[cur ^ 1][rB * 32 + segB]);
      WAIT_ASYNC(3);        // oldest 3 (current tile) complete
    } else {
      WAIT_ASYNC(0);
    }
    __syncthreads();
    v16bf a0 = load_frag(At[cur], rsub0,     lane);
    v16bf a1 = load_frag(At[cur], rsub0 + 1, lane);
    v16bf b0 = load_frag(Bt[cur], csub0,     lane);
    v16bf b1 = load_frag(Bt[cur], csub0 + 1, lane);
    acc00 = __builtin_amdgcn_wmma_f32_16x16x32_bf16(false, a0, false, b0, (short)0, acc00, false, false);
    acc01 = __builtin_amdgcn_wmma_f32_16x16x32_bf16(false, a0, false, b1, (short)0, acc01, false, false);
    acc10 = __builtin_amdgcn_wmma_f32_16x16x32_bf16(false, a1, false, b0, (short)0, acc10, false, false);
    acc11 = __builtin_amdgcn_wmma_f32_16x16x32_bf16(false, a1, false, b1, (short)0, acc11, false, false);
    __syncthreads();        // everyone done reading buf[cur] before overwrite
  }
#else
  // ---------------- fallback: sync staging, single buffer ----------------
  for (int i = 0; i < nb; i++) {
    int k0 = i << 5;
    __syncthreads();
    v4u a0v = {0u,0u,0u,0u}, a1v = {0u,0u,0u,0u};
    if (!aDead) {
      a0v = *(const v4u*)(aRow + k0 + segA);
      a1v = *(const v4u*)(aRow + k0 + segA + 16);
    }
    v4u bv = *(const v4u*)(bRow + k0 + segB);
    *(v4u*)&At[0][rA * 32 + segA]      = a0v;
    *(v4u*)&At[0][rA * 32 + segA + 16] = a1v;
    *(v4u*)&Bt[0][rB * 32 + segB]      = bv;
    __syncthreads();
    v16bf a0 = load_frag(At[0], rsub0,     lane);
    v16bf a1 = load_frag(At[0], rsub0 + 1, lane);
    v16bf b0 = load_frag(Bt[0], csub0,     lane);
    v16bf b1 = load_frag(Bt[0], csub0 + 1, lane);
    acc00 = __builtin_amdgcn_wmma_f32_16x16x32_bf16(false, a0, false, b0, (short)0, acc00, false, false);
    acc01 = __builtin_amdgcn_wmma_f32_16x16x32_bf16(false, a0, false, b1, (short)0, acc01, false, false);
    acc10 = __builtin_amdgcn_wmma_f32_16x16x32_bf16(false, a1, false, b0, (short)0, acc10, false, false);
    acc11 = __builtin_amdgcn_wmma_f32_16x16x32_bf16(false, a1, false, b1, (short)0, acc11, false, false);
  }
#endif

  // ---- epilogue (C/D layout: VGPR r -> M = r + 8*(lane>>4), N = lane&15) ----
  const int mr0 = rsub0 * 16 + (lane >> 4) * 8;
  const int nc0 = colBase + csub0 * 16 + (lane & 15);

  #pragma unroll
  for (int s = 0; s < 4; s++) {                  // subtile (rs, cs)
    const int rs = s >> 1, cs = s & 1;
    v8f& acc = (s == 0) ? acc00 : (s == 1) ? acc01 : (s == 2) ? acc10 : acc11;
    const int ncol = nc0 + cs * 16;
    #pragma unroll
    for (int r = 0; r < 8; r++) {
      int grow = rowBase + mr0 + rs * 16 + r;
      float v  = acc[r];
      if (MODE == 0 || MODE == 2) {
        ((unsigned short*)Cout)[(long)grow * N + ncol] = f2bf(v);
      } else if (MODE == 1) {
        ((float*)Cout)[(long)grow * N + ncol] = v * rowscale[grow];
      } else { // MODE 3
        int tok = perm[grow];
        if (tok >= 0)
          atomicAdd(&((float*)Cout)[(long)tok * HDIM + ncol], v * permw[grow]);
      }
    }
  }
}

// ---------------- router: logits, softmax, top-2, gates, stats ----------------
__global__ void __launch_bounds__(128)
router_kernel(const float* __restrict__ x, const float* __restrict__ gate_w,
              const float* __restrict__ segw,
              int* __restrict__ topk_id, float* __restrict__ topk_w,
              float* __restrict__ gate_sig,
              int* __restrict__ counts, float* __restrict__ probsum)
{
  __shared__ float sW[256 * E_NUM];   // chunk of gate_w, transposed [d][e]
  __shared__ float sG[256];
  __shared__ float bsum[E_NUM];
  __shared__ int   bcnt[E_NUM];

  const int tid = threadIdx.x;
  const int tok = blockIdx.x * 128 + tid;
  const float* xr = x + (long)tok * HDIM;

  if (tid < E_NUM) { bsum[tid] = 0.0f; bcnt[tid] = 0; }

  float acc[E_NUM];
  #pragma unroll
  for (int e = 0; e < E_NUM; e++) acc[e] = 0.0f;
  float accg = 0.0f;

  for (int dc = 0; dc < HDIM; dc += 256) {
    __syncthreads();
    for (int i = tid; i < 256 * E_NUM; i += 128) {
      int d = i >> 4, e = i & 15;
      sW[i] = gate_w[(long)e * HDIM + dc + d];
    }
    for (int i = tid; i < 256; i += 128) sG[i] = segw[dc + i];
    __syncthreads();
    for (int d = 0; d < 256; d += 4) {
      v4f xv = *(const v4f*)(xr + dc + d);
      #pragma unroll
      for (int j = 0; j < 4; j++) {
        float xs = xv[j];
        const float* wrow = sW + (d + j) * E_NUM;
        #pragma unroll
        for (int e = 0; e < E_NUM; e++) acc[e] += xs * wrow[e];
        accg += xs * sG[d + j];
      }
    }
  }

  float m = acc[0];
  #pragma unroll
  for (int e = 1; e < E_NUM; e++) m = fmaxf(m, acc[e]);
  float p[E_NUM], s = 0.0f;
  #pragma unroll
  for (int e = 0; e < E_NUM; e++) { p[e] = __expf(acc[e] - m); s += p[e]; }
  float inv = 1.0f / s;

  int i1 = 0;
  #pragma unroll
  for (int e = 1; e < E_NUM; e++) if (p[e] > p[i1]) i1 = e;
  int i2 = (i1 == 0) ? 1 : 0;
  #pragma unroll
  for (int e = 0; e < E_NUM; e++) if (e != i1 && p[e] > p[i2]) i2 = e;

  float p1 = p[i1] * inv, p2 = p[i2] * inv;
  float wn = 1.0f / (p1 + p2);
  topk_id[tok * 2 + 0] = i1;
  topk_id[tok * 2 + 1] = i2;
  topk_w[tok * 2 + 0]  = p1 * wn;
  topk_w[tok * 2 + 1]  = p2 * wn;
  gate_sig[tok] = 1.0f / (1.0f + __expf(-accg));

  __syncthreads();
  #pragma unroll
  for (int e = 0; e < E_NUM; e++) atomicAdd(&bsum[e], p[e] * inv);
  atomicAdd(&bcnt[i1], 1);
  atomicAdd(&bcnt[i2], 1);
  __syncthreads();
  if (tid < E_NUM) {
    atomicAdd(&probsum[tid], bsum[tid]);
    atomicAdd(&counts[tid],  bcnt[tid]);
  }
}

// ---------------- small helper kernels ----------------
__global__ void init_kernel(int* perm, float* permw, int* counts,
                            float* probsum, int* cursor, unsigned short* zrow) {
  int i = blockIdx.x * 256 + threadIdx.x;
  if (i < MCAP) { perm[i] = -1; permw[i] = 0.0f; }
  if (i < E_NUM) { counts[i] = 0; probsum[i] = 0.0f; cursor[i] = 0; }
  if (i < 128)  zrow[i] = 0;
}

__global__ void finalize_router(const int* __restrict__ counts,
                                const float* __restrict__ probsum,
                                int* __restrict__ po, float* __restrict__ auxdst) {
  if (blockIdx.x == 0 && threadIdx.x == 0) {
    int off = 0;
    float aux = 0.0f;
    for (int e = 0; e < E_NUM; e++) {
      po[e] = off;
      off += (counts[e] + BMROWS - 1) & ~(BMROWS - 1);   // pad to 128 rows
      aux += ((float)counts[e] / (float)(NTOK * TOPKK)) *
             (probsum[e] / (float)NTOK);
    }
    po[E_NUM] = off;
    auxdst[0] = (float)E_NUM * aux;
  }
}

__global__ void scatter_kernel(const int* __restrict__ topk_id,
                               const float* __restrict__ topk_w,
                               const int* __restrict__ po,
                               int* cursor, int* perm, float* permw) {
  int tok = blockIdx.x * 256 + threadIdx.x;
  if (tok >= NTOK) return;
  #pragma unroll
  for (int k = 0; k < TOPKK; k++) {
    int e   = topk_id[tok * 2 + k];
    int pos = po[e] + atomicAdd(&cursor[e], 1);
    perm[pos]  = tok;
    permw[pos] = topk_w[tok * 2 + k];
  }
}

__global__ void cvt_bf16_kernel(const float* __restrict__ in,
                                unsigned short* __restrict__ out, long n) {
  long i = (long)blockIdx.x * blockDim.x + threadIdx.x;
  long st = (long)gridDim.x * blockDim.x;
  for (; i < n; i += st) out[i] = f2bf(in[i]);
}

__global__ void silu_shared_kernel(const unsigned short* __restrict__ g,
                                   const unsigned short* __restrict__ u,
                                   unsigned short* __restrict__ h, long n) {
  long i = (long)blockIdx.x * blockDim.x + threadIdx.x;
  long st = (long)gridDim.x * blockDim.x;
  for (; i < n; i += st)
    h[i] = f2bf(silu_f(bf2f(g[i])) * bf2f(u[i]));
}

__global__ void silu_expert_kernel(const unsigned short* __restrict__ gu,
                                   unsigned short* __restrict__ hx, long n) {
  long i = (long)blockIdx.x * blockDim.x + threadIdx.x;
  long st = (long)gridDim.x * blockDim.x;
  for (; i < n; i += st) {
    long r = i >> 9;            // / IDIM
    long c = i & 511;
    float gg = bf2f(gu[r * (2 * IDIM) + c]);
    float uu = bf2f(gu[r * (2 * IDIM) + IDIM + c]);
    hx[i] = f2bf(silu_f(gg) * uu);
  }
}

// ---------------- launcher ----------------
extern "C" void kernel_launch(void* const* d_in, const int* in_sizes, int n_in,
                              void* d_out, int out_size, void* d_ws, size_t ws_size,
                              hipStream_t stream) {
  const float* x    = (const float*)d_in[0];   // [N, H]
  const float* gw   = (const float*)d_in[1];   // [E, H]
  const float* gup  = (const float*)d_in[2];   // [E, 2I, H]
  const float* dwn  = (const float*)d_in[3];   // [E, H, I]
  const float* sgp  = (const float*)d_in[4];   // [IS, H]
  const float* sup  = (const float*)d_in[5];   // [IS, H]
  const float* sdp  = (const float*)d_in[6];   // [H, IS]
  const float* segw = (const float*)d_in[7];   // [1, H]
  float* out = (float*)d_out;                  // N*H + 1 (aux)

  char* p = (char*)d_ws;
  auto alloc = [&](size_t bytes) -> char* {
    char* r = p;
    p += (bytes + 255) & ~(size_t)255;
    return r;
  };

  unsigned short* Xb  = (unsigned short*)alloc((size_t)NTOK * HDIM * 2);
  unsigned short* Wgu = (unsigned short*)alloc((size_t)E_NUM * 2 * IDIM * HDIM * 2);
  unsigned short* Wd  = (unsigned short*)alloc((size_t)E_NUM * HDIM * IDIM * 2);
  unsigned short* Wsg = (unsigned short*)alloc((size_t)ISDIM * HDIM * 2);
  unsigned short* Wsu = (unsigned short*)alloc((size_t)ISDIM * HDIM * 2);
  unsigned short* Wsd = (unsigned short*)alloc((size_t)HDIM * ISDIM * 2);

  size_t sharedBytes = 3ull * NTOK * ISDIM * 2;
  size_t expertBytes = (size_t)MCAP * 2 * IDIM * 2 + (size_t)MCAP * IDIM * 2;
  char* region = alloc(sharedBytes > expertBytes ? sharedBytes : expertBytes);
  unsigned short* gs = (unsigned short*)region;
  unsigned short* us = gs + (size_t)NTOK * ISDIM;
  unsigned short* Hs = us + (size_t)NTOK * ISDIM;
  unsigned short* gu = (unsigned short*)region;                 // reuse after S2
  unsigned short* Hx = gu + (size_t)MCAP * 2 * IDIM;

  int*   topk_id  = (int*)  alloc((size_t)NTOK * 2 * 4);
  float* topk_w   = (float*)alloc((size_t)NTOK * 2 * 4);
  float* gate_sig = (float*)alloc((size_t)NTOK * 4);
  int*   counts   = (int*)  alloc(E_NUM * 4);
  float* probsum  = (float*)alloc(E_NUM * 4);
  int*   cursor   = (int*)  alloc(E_NUM * 4);
  int*   po       = (int*)  alloc((E_NUM + 1) * 4);
  int*   perm     = (int*)  alloc((size_t)MCAP * 4);
  float* permw    = (float*)alloc((size_t)MCAP * 4);
  unsigned short* zrow = (unsigned short*)alloc(256);

  // 1) init + routing
  init_kernel<<<(MCAP + 255) / 256, 256, 0, stream>>>(perm, permw, counts,
                                                      probsum, cursor, zrow);
  router_kernel<<<NTOK / 128, 128, 0, stream>>>(x, gw, segw, topk_id, topk_w,
                                                gate_sig, counts, probsum);
  finalize_router<<<1, 32, 0, stream>>>(counts, probsum, po, out + (size_t)NTOK * HDIM);
  scatter_kernel<<<NTOK / 256, 256, 0, stream>>>(topk_id, topk_w, po, cursor, perm, permw);

  // 2) f32 -> bf16 operand conversion
  cvt_bf16_kernel<<<2048, 256, 0, stream>>>(x,   Xb,  (long)NTOK * HDIM);
  cvt_bf16_kernel<<<4096, 256, 0, stream>>>(gup, Wgu, (long)E_NUM * 2 * IDIM * HDIM);
  cvt_bf16_kernel<<<2048, 256, 0, stream>>>(dwn, Wd,  (long)E_NUM * HDIM * IDIM);
  cvt_bf16_kernel<<<1024, 256, 0, stream>>>(sgp, Wsg, (long)ISDIM * HDIM);
  cvt_bf16_kernel<<<1024, 256, 0, stream>>>(sup, Wsu, (long)ISDIM * HDIM);
  cvt_bf16_kernel<<<1024, 256, 0, stream>>>(sdp, Wsd, (long)HDIM * ISDIM);

  // 3) shared expert path (initializes d_out with gated shared output)
  dim3 gS1(NTOK / BMROWS, ISDIM / 64);
  gemm_bf16<0><<<gS1, 256, 0, stream>>>(Xb, Wsg, gs, ISDIM, HDIM,
                                        nullptr, nullptr, nullptr, nullptr, zrow, 0);
  gemm_bf16<0><<<gS1, 256, 0, stream>>>(Xb, Wsu, us, ISDIM, HDIM,
                                        nullptr, nullptr, nullptr, nullptr, zrow, 0);
  silu_shared_kernel<<<4096, 256, 0, stream>>>(gs, us, Hs, (long)NTOK * ISDIM);
  dim3 gS2(NTOK / BMROWS, HDIM / 64);
  gemm_bf16<1><<<gS2, 256, 0, stream>>>(Hs, Wsd, out, HDIM, ISDIM,
                                        nullptr, nullptr, nullptr, gate_sig, zrow, 0);

  // 4) routed experts (top-2 sparse, gathered rows, fixed max grid + device early-exit)
  dim3 gE1(MAXTILES, (2 * IDIM) / 64);
  gemm_bf16<2><<<gE1, 256, 0, stream>>>(Xb, Wgu, gu, 2 * IDIM, HDIM,
                                        perm, nullptr, po, nullptr, zrow,
                                        (long)2 * IDIM * HDIM);
  silu_expert_kernel<<<4096, 256, 0, stream>>>(gu, Hx, (long)MCAP * IDIM);
  dim3 gE2(MAXTILES, HDIM / 64);
  gemm_bf16<3><<<gE2, 256, 0, stream>>>(Hx, Wd, out, HDIM, IDIM,
                                        perm, permw, po, nullptr, zrow,
                                        (long)HDIM * IDIM);
}
